// Basic_RSNN_BRP_88845693485071
// MI455X (gfx1250) — matrix operand order repaired and checked by reading, not compiled, not run on hardware
//
#include <hip/hip_runtime.h>
#include <math.h>

// ---------------- problem constants ----------------
#define B_SZ    128
#define T_SZ    500
#define NIN     784
#define H_SZ    1024
#define NSLICE  102          // 1024 / 10
#define KT_FC1  25           // ceil(784/32): K padded 784 -> 800
#define KT_REC  32           // 1024/32
#define NT      (H_SZ / 16)  // 64 N-tiles
#define NFRAG   (KT_FC1 + KT_REC)   // 57 B-fragments per ntile
#define NBLK    64           // persistent grid: one block per ntile
#define TPB     256          // 8 wave32 = 8 mtiles per block

// ---------------- CDNA5 WMMA types ----------------
typedef __attribute__((ext_vector_type(16))) __bf16       v16bf;
typedef __attribute__((ext_vector_type(8)))  float        v8f;
typedef __attribute__((ext_vector_type(4)))  unsigned int u32x4;
typedef __attribute__((ext_vector_type(4)))  float        f32x4;

union Frag {
  u32x4          q[2];   // two 16B chunks -> b128 loads
  v16bf          v;      // WMMA operand view
  unsigned short s[16];  // bf16 bit-pattern view
};

__device__ __forceinline__ unsigned short f2bf(float f) {
  union { float f; unsigned int u; } c; c.f = f;
  unsigned int u = c.u;
  unsigned int r = (u + 0x7FFFu + ((u >> 16) & 1u)) >> 16;  // RNE
  return (unsigned short)r;
}

// One 16x16x32 bf16 WMMA: A from two 16B chunks (a0p, a1p), B from one
// contiguous 32B fragment (bp, LDS), accumulate into c.
__device__ __forceinline__ v8f mm_frag(const unsigned short* a0p,
                                       const unsigned short* a1p,
                                       const unsigned short* bp, v8f c) {
  Frag a, b;
  a.q[0] = *reinterpret_cast<const u32x4*>(a0p);
  a.q[1] = *reinterpret_cast<const u32x4*>(a1p);
  const u32x4* bq = reinterpret_cast<const u32x4*>(bp);
  b.q[0] = bq[0];
  b.q[1] = bq[1];
  return __builtin_amdgcn_wmma_f32_16x16x32_bf16(false, a.v, false, b.v,
                                                 (short)0, c, false, false);
}

// ---------------------------------------------------------------------------
// Pre-pass 1: repack fp32 [K, H] weights into per-lane WMMA B-fragment layout
// (bf16). Tile (kt, nt), lane L: column nt*16+(L&15), K=(L>>4)*8+e+(e&8), e=0..15,
// stored as 32 contiguous bytes per lane.
// ---------------------------------------------------------------------------
__global__ void pack_b_kernel(const float* __restrict__ src,
                              unsigned short* __restrict__ dst,
                              int K, int KT) {
  int idx = blockIdx.x * blockDim.x + threadIdx.x;    // (kt, nt, lane)
  int total = KT * NT * 32;
  if (idx >= total) return;
  int lane = idx & 31;
  int nt   = (idx >> 5) & (NT - 1);
  int kt   = idx / (NT * 32);
  int col   = nt * 16 + (lane & 15);
  int kbase = kt * 32 + (lane >> 4) * 8;
  Frag f;
#pragma unroll
  for (int e = 0; e < 16; ++e) {
    int k = kbase + e + (e & 8);
    f.s[e] = (k < K) ? f2bf(src[(size_t)k * H_SZ + col]) : (unsigned short)0;
  }
  u32x4* out = reinterpret_cast<u32x4*>(dst + (size_t)idx * 16);
  out[0] = f.q[0];
  out[1] = f.q[1];
}

// ---------------------------------------------------------------------------
// Pre-pass 2: pack x [B, T, NIN] fp32 into bf16 A-fragment layout
// xp[t][mtile][kt][lane][16]. Fully parallel; removes ALL fp32->bf16 VALU
// (and the 64x redundant conversion) from the sequential scan.
// ---------------------------------------------------------------------------
__global__ void pack_x_kernel(const float* __restrict__ x,
                              unsigned short* __restrict__ xp) {
  int idx = blockIdx.x * blockDim.x + threadIdx.x;   // (t, mt, kt, lane)
  const int total = T_SZ * 8 * KT_FC1 * 32;          // 3,200,000
  if (idx >= total) return;
  int lane = idx & 31;
  int kt   = (idx >> 5) % KT_FC1;
  int mt   = (idx / (32 * KT_FC1)) & 7;
  int t    = idx / (32 * KT_FC1 * 8);
  int lhi  = lane >> 4;
  int row  = mt * 16 + (lane & 15);
  int k0   = kt * 32;
  const float* xrow = x + ((size_t)row * T_SZ + t) * NIN;
  Frag f;
  f32x4 x0 = *reinterpret_cast<const f32x4*>(xrow + k0 + lhi * 8);
  f32x4 x1 = *reinterpret_cast<const f32x4*>(xrow + k0 + lhi * 8 + 4);
#pragma unroll
  for (int e = 0; e < 4; ++e) { f.s[e] = f2bf(x0[e]); f.s[4 + e] = f2bf(x1[e]); }
  if (k0 + 16 < NIN) {   // kt==24 upper half is the K=784->800 zero pad
    f32x4 x2 = *reinterpret_cast<const f32x4*>(xrow + k0 + 16 + lhi * 8);
    f32x4 x3 = *reinterpret_cast<const f32x4*>(xrow + k0 + 16 + lhi * 8 + 4);
#pragma unroll
    for (int e = 0; e < 4; ++e) { f.s[8 + e] = f2bf(x2[e]); f.s[12 + e] = f2bf(x3[e]); }
  } else {
#pragma unroll
    for (int e = 8; e < 16; ++e) f.s[e] = 0;
  }
  u32x4* out = reinterpret_cast<u32x4*>(xp + (size_t)idx * 16);
  out[0] = f.q[0];
  out[1] = f.q[1];
}

// Zero membrane ([H,B] fp32), t=0 spikes ([B,H] bf16), and barrier state.
__global__ void zero_state_kernel(float* __restrict__ mem,
                                  unsigned short* __restrict__ spk,
                                  unsigned int* __restrict__ arrive,
                                  unsigned int* __restrict__ release_) {
  int i = blockIdx.x * blockDim.x + threadIdx.x;
  if (i < B_SZ * H_SZ) { mem[i] = 0.0f; spk[i] = 0; }
  if (i < NBLK) arrive[i] = 0u;
  if (i == NBLK) *release_ = 0u;
}

// ---------------------------------------------------------------------------
// Persistent scan kernel: ONE launch for all 500 steps.
//   grid = 64 blocks (block = ntile), 8 wave32s per block (wave = mtile).
// Prologue: async-copy this ntile's 57 B-fragments (58,368 B) into LDS with
//           global_load_async_to_lds_b128 + s_wait_asynccnt. The smem pointer
//           is passed INTO the asm so the allocation escapes and the "memory"
//           clobber pins the later ds_load_b128 reads (round-2 bug: offsets
//           alone let LLVM fold the never-"stored" LDS reads to undef).
// Per step: 57 WMMAs spread over 4 independent accumulators (covers the
//           5-slot bf16 WMMA->WMMA RAW hazard window without v_nop padding),
//           LIF pointwise update, then a distributed grid barrier.
// ---------------------------------------------------------------------------
__global__ void __launch_bounds__(TPB, 1)
rsnn_persist_kernel(const unsigned short* __restrict__ xp,    // packed x A-frags
                    const unsigned short* __restrict__ fc1p,  // packed B-frags
                    const unsigned short* __restrict__ recp,  // packed B-frags
                    float* __restrict__ mem,                  // [H, B] fp32
                    unsigned short* __restrict__ spkA,        // [B, H] bf16
                    unsigned short* __restrict__ spkB,        // [B, H] bf16
                    float* __restrict__ out,                  // [B, T, NSLICE]
                    unsigned int* __restrict__ arrive,        // [NBLK]
                    unsigned int* __restrict__ release_,      // [1]
                    float decay) {
  __shared__ __align__(16) unsigned short smem[NFRAG * 512];  // 58,368 B

  const int ntile = blockIdx.x;          // 0..63
  const int wid   = threadIdx.x >> 5;    // mtile 0..7
  const int lane  = threadIdx.x & 31;
  const int lrow  = lane & 15;
  const int lhi   = lane >> 4;
  const int m0    = wid * 16;

  // ---- prologue: B fragments -> LDS via async global->LDS DMA ----
  // smem is the only LDS allocation -> instruction's LDS byte address is the
  // array offset. The pointer operand only exists to escape the allocation.
  {
    unsigned long long smem_escape = (unsigned long long)(size_t)(void*)smem;
    for (int c = threadIdx.x; c < NFRAG * 64; c += TPB) { // 3648 x 16B chunks
      int f = c >> 6;        // fragment index (64 x 16B chunks per 1KB frag)
      int w = c & 63;
      const unsigned short* src = (f < KT_FC1)
          ? (fc1p + (((size_t)f * NT + ntile) * 32) * 16 + w * 8)
          : (recp + (((size_t)(f - KT_FC1) * NT + ntile) * 32) * 16 + w * 8);
      unsigned lds_off = (unsigned)(f * 1024 + w * 16);
      asm volatile("global_load_async_to_lds_b128 %0, %1, off"
                   :: "v"(lds_off), "v"((unsigned long long)(size_t)src),
                      "v"(smem_escape)
                   : "memory");
    }
    asm volatile("s_wait_asynccnt 0" ::: "memory");
  }
  __syncthreads();

  // Epilogue addressing (C layout: VGPR r -> M = r + 8*lhi, N = lrow).
  const int h     = ntile * 16 + lrow;
  const int bbase = m0 + lhi * 8;
  float* mrow = mem + (size_t)h * B_SZ + bbase;

  for (int t = 0; t < T_SZ; ++t) {
    const unsigned short* spk_in  = (t & 1) ? spkB : spkA;
    unsigned short*       spk_out = (t & 1) ? spkA : spkB;

    v8f c0 = {0.f, 0.f, 0.f, 0.f, 0.f, 0.f, 0.f, 0.f};
    v8f c1 = c0, c2 = c0, c3 = c0;

    // ---- input current: prepacked bf16 x_t A-frags  @  fc1 B-frags (LDS) ----
    // xp frag (t, wid, kt): 32B per lane, contiguous; kt stride = 1024 B.
    const unsigned short* xbase =
        xp + (((size_t)t * 8 + wid) * KT_FC1 * 32) * 16 + lane * 16;
    const unsigned short* bfc = &smem[lane * 16];
#pragma unroll
    for (int kt = 0; kt < 24; kt += 4) {   // 4 independent accumulator chains
      c0 = mm_frag(xbase + (kt + 0) * 512, xbase + (kt + 0) * 512 + 8,
                   bfc + (kt + 0) * 512, c0);
      c1 = mm_frag(xbase + (kt + 1) * 512, xbase + (kt + 1) * 512 + 8,
                   bfc + (kt + 1) * 512, c1);
      c2 = mm_frag(xbase + (kt + 2) * 512, xbase + (kt + 2) * 512 + 8,
                   bfc + (kt + 2) * 512, c2);
      c3 = mm_frag(xbase + (kt + 3) * 512, xbase + (kt + 3) * 512 + 8,
                   bfc + (kt + 3) * 512, c3);
    }
    c0 = mm_frag(xbase + 24 * 512, xbase + 24 * 512 + 8, bfc + 24 * 512, c0);

    // ---- recurrent current: spk_{t-1} A-frags  @  W_rec B-frags (LDS) ----
    // spk A-frag chunks: srow + kt*32 + lhi*8 and +16 (32B apart).
    const unsigned short* srow = spk_in + (size_t)(m0 + lrow) * H_SZ + lhi * 8;
    const unsigned short* brc  = &smem[(KT_FC1 * 32 + lane) * 16];
#pragma unroll
    for (int kt = 0; kt < KT_REC; kt += 4) {
      c1 = mm_frag(srow + (kt + 0) * 32, srow + (kt + 0) * 32 + 16,
                   brc + (kt + 0) * 512, c1);
      c2 = mm_frag(srow + (kt + 1) * 32, srow + (kt + 1) * 32 + 16,
                   brc + (kt + 1) * 512, c2);
      c3 = mm_frag(srow + (kt + 2) * 32, srow + (kt + 2) * 32 + 16,
                   brc + (kt + 2) * 512, c3);
      c0 = mm_frag(srow + (kt + 3) * 32, srow + (kt + 3) * 32 + 16,
                   brc + (kt + 3) * 512, c0);
    }
    v8f acc = (c0 + c1) + (c2 + c3);

    // ---- LIF pointwise update + fire + emit slice ----
    f32x4 mv0 = *reinterpret_cast<const f32x4*>(mrow);
    f32x4 mv1 = *reinterpret_cast<const f32x4*>(mrow + 4);
    float nm[8];
#pragma unroll
    for (int r = 0; r < 8; ++r) {
      const int bb = bbase + r;
      const float sp = spk_in[(size_t)bb * H_SZ + h] ? 1.0f : 0.0f;
      const float m  = (r < 4) ? mv0[r] : mv1[r - 4];
      const float v  = m * decay * (1.0f - sp) + acc[r];
      nm[r] = v;
      const float ns = (v >= 1.0f) ? 1.0f : 0.0f;
      spk_out[(size_t)bb * H_SZ + h] =
          (ns != 0.0f) ? (unsigned short)0x3F80 : (unsigned short)0;
      if (h < NSLICE)
        out[((size_t)bb * T_SZ + t) * NSLICE + h] = ns;
    }
    f32x4 w0 = {nm[0], nm[1], nm[2], nm[3]};
    f32x4 w1 = {nm[4], nm[5], nm[6], nm[7]};
    *reinterpret_cast<f32x4*>(mrow)     = w0;
    *reinterpret_cast<f32x4*>(mrow + 4) = w1;

    // ---- distributed grid barrier (monotone counters, no resets) ----
    __threadfence();            // spk_out / mem visible device-wide
    __syncthreads();
    if (threadIdx.x == 0)
      __hip_atomic_store(&arrive[blockIdx.x], (unsigned)(t + 1),
                         __ATOMIC_RELEASE, __HIP_MEMORY_SCOPE_AGENT);
    if (blockIdx.x == 0) {      // block-uniform branch
      if (threadIdx.x < NBLK) {
        while (__hip_atomic_load(&arrive[threadIdx.x], __ATOMIC_ACQUIRE,
                                 __HIP_MEMORY_SCOPE_AGENT) < (unsigned)(t + 1))
          __builtin_amdgcn_s_sleep(2);
      }
      __syncthreads();
      if (threadIdx.x == 0)
        __hip_atomic_store(release_, (unsigned)(t + 1),
                           __ATOMIC_RELEASE, __HIP_MEMORY_SCOPE_AGENT);
    }
    if (threadIdx.x == 0) {
      while (__hip_atomic_load(release_, __ATOMIC_ACQUIRE,
                               __HIP_MEMORY_SCOPE_AGENT) < (unsigned)(t + 1))
        __builtin_amdgcn_s_sleep(2);
    }
    __syncthreads();
    __threadfence();
  }
}

// ---------------- workspace layout (bytes, 16B-aligned) ----------------
#define WS_FC1P  ((size_t)0)                     // 1,638,400 B
#define WS_RECP  ((size_t)1638400)               // 2,097,152 B
#define WS_MEM   (WS_RECP + (size_t)2097152)     //   524,288 B
#define WS_SPKA  (WS_MEM  + (size_t)524288)      //   262,144 B
#define WS_SPKB  (WS_SPKA + (size_t)262144)      //   262,144 B
#define WS_SYNC  (WS_SPKB + (size_t)262144)      //       512 B (arrive+release)
#define WS_XP    (WS_SYNC + (size_t)512)         // 102,400,000 B packed x
// total ~ 107.2 MB (inputs alone are 206 MB; ws assumed comparable)

extern "C" void kernel_launch(void* const* d_in, const int* in_sizes, int n_in,
                              void* d_out, int out_size, void* d_ws, size_t ws_size,
                              hipStream_t stream) {
  (void)in_sizes; (void)n_in; (void)out_size; (void)ws_size;
  const float* x   = (const float*)d_in[0];   // [128, 500, 784]
  const float* fc1 = (const float*)d_in[1];   // [784, 1024]
  const float* rec = (const float*)d_in[2];   // [1024, 1024]
  float* out = (float*)d_out;                 // [128, 500, 102]

  char* ws = (char*)d_ws;
  unsigned short* fc1p    = (unsigned short*)(ws + WS_FC1P);
  unsigned short* recp    = (unsigned short*)(ws + WS_RECP);
  float*          memb    = (float*)(ws + WS_MEM);
  unsigned short* spkA    = (unsigned short*)(ws + WS_SPKA);
  unsigned short* spkB    = (unsigned short*)(ws + WS_SPKB);
  unsigned int*   arrive  = (unsigned int*)(ws + WS_SYNC);
  unsigned int*   release_= (unsigned int*)(ws + WS_SYNC + 256);
  unsigned short* xpk     = (unsigned short*)(ws + WS_XP);

  // Pre-passes (fully parallel, off the critical scan path).
  {
    int th1 = KT_FC1 * NT * 32;                 // 51,200
    int th2 = KT_REC * NT * 32;                 // 65,536
    int thx = T_SZ * 8 * KT_FC1 * 32;           // 3,200,000
    pack_b_kernel<<<(th1 + 255) / 256, 256, 0, stream>>>(fc1, fc1p, NIN, KT_FC1);
    pack_b_kernel<<<(th2 + 255) / 256, 256, 0, stream>>>(rec, recp, H_SZ, KT_REC);
    pack_x_kernel<<<(thx + 255) / 256, 256, 0, stream>>>(x, xpk);
    zero_state_kernel<<<(B_SZ * H_SZ + 255) / 256, 256, 0, stream>>>(
        memb, spkA, arrive, release_);
  }

  const float decay = 1.0f / (1.0f + expf(-2.0f));  // sigmoid(INIT_TAU)

  // Single persistent launch for the whole 500-step scan.
  rsnn_persist_kernel<<<NBLK, TPB, 0, stream>>>(xpk, fc1p, recp, memb, spkA,
                                                spkB, out, arrive, release_,
                                                decay);
}
